// MoE_50319836840186
// MI455X (gfx1250) — compile-verified, matrix operand order checked by reading, not run on hardware
//
#include <hip/hip_runtime.h>

// ================= MI455X (gfx1250) MoE FFN =================
// Routed top-2 MoE: 137 GFLOP compute vs 284 MB compulsory bytes (12.2us @
// 23.3 TB/s) -> compute-bound -> f16 WMMA (16x16x32, f32 accum) + routing
// (4x fewer FLOPs than dense). Each expert's 16MB weight slab fits the 192MB
// L2, so tile re-reads don't touch HBM. wave32, 8 waves/block, 128x128 block
// tile, 2x4 wave grid, 4x2 WMMA accum tiles per wave.
// Round-1 fix: branch-free vectorized staging (b128 global loads, b64/b128
// LDS stores) + register double-buffering so global latency overlaps WMMA.

typedef __attribute__((ext_vector_type(16))) _Float16 v16h;
typedef __attribute__((ext_vector_type(8)))  _Float16 v8h;
typedef __attribute__((ext_vector_type(4)))  _Float16 v4h;
typedef __attribute__((ext_vector_type(8)))  float    v8f;

union F16Frag { v16h h; uint4 q[2]; };

constexpr int N_TOKENS  = 4096;
constexpr int N_EMBD    = 1024;
constexpr int N_HID     = 4096;   // 4 * N_EMBD
constexpr int N_EXP     = 8;
constexpr int MAX_SLOTS = 2 * N_TOKENS;
constexpr int LDP       = 40;     // padded LDS row stride (halves): 80B, 16B-aligned, bank-spread

// ---------------- gating: one wave per token ----------------
__global__ __launch_bounds__(256) void moe_gate(const float* __restrict__ x,
                                                const float* __restrict__ gw,
                                                int* __restrict__ eidx,
                                                float* __restrict__ ewt)
{
    const int wid  = threadIdx.x >> 5;
    const int lane = threadIdx.x & 31;
    const int n    = blockIdx.x * 8 + wid;
    const float* xr = x + (size_t)n * N_EMBD;

    float acc[N_EXP];
#pragma unroll
    for (int e = 0; e < N_EXP; ++e) acc[e] = 0.f;

    for (int c = lane; c < N_EMBD; c += 32) {
        float xv = xr[c];
        const float* g = gw + c * N_EXP;
#pragma unroll
        for (int e = 0; e < N_EXP; ++e) acc[e] += xv * g[e];
    }
#pragma unroll
    for (int e = 0; e < N_EXP; ++e)
        for (int off = 16; off > 0; off >>= 1)
            acc[e] += __shfl_xor(acc[e], off);

    if (lane == 0) {
        int i1 = 0; float l1 = acc[0];
        for (int e = 1; e < N_EXP; ++e) if (acc[e] > l1) { l1 = acc[e]; i1 = e; }
        int i2 = (i1 == 0) ? 1 : 0; float l2 = acc[i2];
        for (int e = 0; e < N_EXP; ++e)
            if (e != i1 && acc[e] > l2) { l2 = acc[e]; i2 = e; }
        // renormalized top-2 softmax == softmax over {l1,l2}
        float p1 = 1.f / (1.f + expf(l2 - l1));
        eidx[2 * n]     = i1;
        eidx[2 * n + 1] = i2;
        ewt[2 * n]      = p1;
        ewt[2 * n + 1]  = 1.f - p1;
    }
}

// ---------------- routing bookkeeping ----------------
__global__ void moe_zero(int* __restrict__ counts)
{
    if (threadIdx.x < N_EXP) counts[threadIdx.x] = 0;
}

__global__ __launch_bounds__(256) void moe_count(const int* __restrict__ eidx,
                                                 int* __restrict__ counts)
{
    int n = blockIdx.x * 256 + threadIdx.x;
    if (n < N_TOKENS) {
        atomicAdd(&counts[eidx[2 * n]], 1);
        atomicAdd(&counts[eidx[2 * n + 1]], 1);
    }
}

__global__ void moe_scan(const int* __restrict__ counts,
                         int* __restrict__ offs,
                         int* __restrict__ cursor)
{
    if (threadIdx.x == 0 && blockIdx.x == 0) {
        int acc = 0;
        for (int e = 0; e < N_EXP; ++e) {
            offs[e]   = acc;
            cursor[e] = acc;
            acc += counts[e];
        }
        offs[N_EXP] = acc;
    }
}

__global__ __launch_bounds__(256) void moe_scatter(const int* __restrict__ eidx,
                                                   int* __restrict__ cursor,
                                                   int* __restrict__ btok,
                                                   int* __restrict__ slots)
{
    int n = blockIdx.x * 256 + threadIdx.x;
    if (n < N_TOKENS) {
#pragma unroll
        for (int k = 0; k < 2; ++k) {
            int e = eidx[2 * n + k];
            int p = atomicAdd(&cursor[e], 1);
            btok[p] = n;
            slots[2 * n + k] = p;
        }
    }
}

// ---------------- FFN stage 1: H = gelu(Xg @ W1 + b1), f16 out ----------------
__global__ __launch_bounds__(256) void moe_ffn1(const float* __restrict__ x,
                                                const float* __restrict__ w1,
                                                const float* __restrict__ b1,
                                                const int* __restrict__ counts,
                                                const int* __restrict__ offs,
                                                const int* __restrict__ btok,
                                                _Float16* __restrict__ H)
{
    const int e   = blockIdx.z;
    const int cnt = counts[e];
    const int rowbase = blockIdx.y * 128;
    if (rowbase >= cnt) return;
    int rowsv = cnt - rowbase; if (rowsv > 128) rowsv = 128;
    const int colbase  = blockIdx.x * 128;
    const int slotbase = offs[e] + rowbase;
    const float* W = w1 + (size_t)e * N_EMBD * N_HID;
    const float* B = b1 + (size_t)e * N_HID;

    __shared__ _Float16 sA[128 * LDP];
    __shared__ _Float16 sB[128 * LDP];
    __shared__ int tok[128];

    const int tid = threadIdx.x;
    if (tid < 128) tok[tid] = (tid < rowsv) ? btok[slotbase + tid] : btok[slotbase];

    const int wid  = tid >> 5, lane = tid & 31;
    const int wr   = (wid >> 2) * 64;     // wave row base within tile
    const int wc   = (wid & 3) * 32;      // wave col base within tile
    const int l15  = lane & 15;
    const int kro  = (lane >> 4) * 8;     // A/B-frag K chunk (ISA 16-bit layout)
    const int mlo  = (lane >> 4) * 8;     // C/D-frag row offset

    v8f acc[4][2];
#pragma unroll
    for (int mt = 0; mt < 4; ++mt)
#pragma unroll
        for (int nt = 0; nt < 2; ++nt)
#pragma unroll
            for (int i = 0; i < 8; ++i) acc[mt][nt][i] = 0.f;

    __syncthreads();   // tok visible

    // ---- hoisted, branch-free staging pointers (clamped rows) ----
    const float* ap[4];   _Float16* sap[4];            // A: 4x float4 per thread
    const float* bp[4];   _Float16* sbp[4];            // B: 4x float4 per thread (transposed scatter)
#pragma unroll
    for (int i = 0; i < 4; ++i) {
        int id = tid + i * 256;
        int r  = id >> 3;
        int k0 = (id & 7) * 4;
        int rc = (r < rowsv) ? r : (rowsv - 1);        // clamp: padded rows read valid data, never stored
        ap[i]  = x + (size_t)tok[rc] * N_EMBD + k0;
        sap[i] = sA + r * LDP + k0;
    }
#pragma unroll
    for (int i = 0; i < 4; ++i) {
        int id = tid + i * 256;
        int k  = id >> 5;
        int n0 = (id & 31) * 4;
        bp[i]  = W + (size_t)k * N_HID + colbase + n0;
        sbp[i] = sB + n0 * LDP + k;
    }

    // ---- software pipeline: regs hold next tile while WMMA runs on current ----
    float4 ra[4], rb[4];
#pragma unroll
    for (int i = 0; i < 4; ++i) {
        ra[i] = *reinterpret_cast<const float4*>(ap[i]);
        rb[i] = *reinterpret_cast<const float4*>(bp[i]);
    }

    for (int kk = 0; kk < N_EMBD; kk += 32) {
        __syncthreads();                                // frags of prev step consumed
#pragma unroll
        for (int i = 0; i < 4; ++i) {                   // A tile -> LDS (pk converts + b64 store)
            v4h h;
            h.x = (_Float16)ra[i].x; h.y = (_Float16)ra[i].y;
            h.z = (_Float16)ra[i].z; h.w = (_Float16)ra[i].w;
            *reinterpret_cast<v4h*>(sap[i]) = h;
        }
#pragma unroll
        for (int i = 0; i < 4; ++i) {                   // B tile -> LDS transposed
            sbp[i][0 * LDP] = (_Float16)rb[i].x;
            sbp[i][1 * LDP] = (_Float16)rb[i].y;
            sbp[i][2 * LDP] = (_Float16)rb[i].z;
            sbp[i][3 * LDP] = (_Float16)rb[i].w;
        }
        __syncthreads();

        if (kk + 32 < N_EMBD) {                         // issue next tile's loads early
#pragma unroll
            for (int i = 0; i < 4; ++i) {
                ap[i] += 32;
                ra[i] = *reinterpret_cast<const float4*>(ap[i]);
            }
#pragma unroll
            for (int i = 0; i < 4; ++i) {
                bp[i] += (size_t)32 * N_HID;
                rb[i] = *reinterpret_cast<const float4*>(bp[i]);
            }
            if (kk + 64 < N_EMBD)                       // warm L2 one more tile ahead
                __builtin_prefetch(bp[tid & 3] + (size_t)32 * N_HID, 0, 1);
        }

        F16Frag af[4], bf[2];
#pragma unroll
        for (int mt = 0; mt < 4; ++mt) {
            const _Float16* p = &sA[(wr + mt * 16 + l15) * LDP + kro];
            af[mt].q[0] = *(const uint4*)p;
            af[mt].q[1] = *(const uint4*)(p + 16);
        }
#pragma unroll
        for (int nt = 0; nt < 2; ++nt) {
            const _Float16* p = &sB[(wc + nt * 16 + l15) * LDP + kro];
            bf[nt].q[0] = *(const uint4*)p;
            bf[nt].q[1] = *(const uint4*)(p + 16);
        }
#pragma unroll
        for (int mt = 0; mt < 4; ++mt)
#pragma unroll
            for (int nt = 0; nt < 2; ++nt)
                acc[mt][nt] = __builtin_amdgcn_wmma_f32_16x16x32_f16(
                    false, af[mt].h, false, bf[nt].h, (short)0, acc[mt][nt], false, false);
    }

    // epilogue: bias + exact gelu, store f16
#pragma unroll
    for (int mt = 0; mt < 4; ++mt)
#pragma unroll
        for (int nt = 0; nt < 2; ++nt) {
            int col = colbase + wc + nt * 16 + l15;
            float bias = B[col];
#pragma unroll
            for (int i = 0; i < 8; ++i) {
                int row = wr + mt * 16 + mlo + i;
                if (row < rowsv) {
                    float v = acc[mt][nt][i] + bias;
                    float g = 0.5f * v * (1.f + erff(v * 0.70710678118654752f));
                    H[(size_t)(slotbase + row) * N_HID + col] = (_Float16)g;
                }
            }
        }
}

// ---------------- FFN stage 2: Y = H @ W2 + b2, f32 out per slot ----------------
__global__ __launch_bounds__(256) void moe_ffn2(const _Float16* __restrict__ H,
                                                const float* __restrict__ w2,
                                                const float* __restrict__ b2,
                                                const int* __restrict__ counts,
                                                const int* __restrict__ offs,
                                                float* __restrict__ Yg)
{
    const int e   = blockIdx.z;
    const int cnt = counts[e];
    const int rowbase = blockIdx.y * 128;
    if (rowbase >= cnt) return;
    int rowsv = cnt - rowbase; if (rowsv > 128) rowsv = 128;
    const int colbase  = blockIdx.x * 128;
    const int slotbase = offs[e] + rowbase;
    const float* W = w2 + (size_t)e * N_HID * N_EMBD;
    const float* B = b2 + (size_t)e * N_EMBD;

    __shared__ _Float16 sA[128 * LDP];
    __shared__ _Float16 sB[128 * LDP];

    const int tid  = threadIdx.x;
    const int wid  = tid >> 5, lane = tid & 31;
    const int wr   = (wid >> 2) * 64;
    const int wc   = (wid & 3) * 32;
    const int l15  = lane & 15;
    const int kro  = (lane >> 4) * 8;
    const int mlo  = (lane >> 4) * 8;

    v8f acc[4][2];
#pragma unroll
    for (int mt = 0; mt < 4; ++mt)
#pragma unroll
        for (int nt = 0; nt < 2; ++nt)
#pragma unroll
            for (int i = 0; i < 8; ++i) acc[mt][nt][i] = 0.f;

    // ---- hoisted staging pointers ----
    const _Float16* ap[2];  _Float16* sap[2];          // A: 2x v8h (f16 direct) per thread
    const float*    bp[4];  _Float16* sbp[4];          // B: 4x float4 transposed scatter
#pragma unroll
    for (int i = 0; i < 2; ++i) {
        int id = tid + i * 256;
        int r  = id >> 2;
        int k0 = (id & 3) * 8;
        int rc = (r < rowsv) ? r : (rowsv - 1);
        ap[i]  = H + (size_t)(slotbase + rc) * N_HID + k0;
        sap[i] = sA + r * LDP + k0;
    }
#pragma unroll
    for (int i = 0; i < 4; ++i) {
        int id = tid + i * 256;
        int k  = id >> 5;
        int n0 = (id & 31) * 4;
        bp[i]  = W + (size_t)k * N_EMBD + colbase + n0;
        sbp[i] = sB + n0 * LDP + k;
    }

    v8h   ra[2];
    float4 rb[4];
#pragma unroll
    for (int i = 0; i < 2; ++i) ra[i] = *reinterpret_cast<const v8h*>(ap[i]);
#pragma unroll
    for (int i = 0; i < 4; ++i) rb[i] = *reinterpret_cast<const float4*>(bp[i]);

    for (int kk = 0; kk < N_HID; kk += 32) {
        __syncthreads();
#pragma unroll
        for (int i = 0; i < 2; ++i)
            *reinterpret_cast<v8h*>(sap[i]) = ra[i];    // b128 LDS store, no converts
#pragma unroll
        for (int i = 0; i < 4; ++i) {
            sbp[i][0 * LDP] = (_Float16)rb[i].x;
            sbp[i][1 * LDP] = (_Float16)rb[i].y;
            sbp[i][2 * LDP] = (_Float16)rb[i].z;
            sbp[i][3 * LDP] = (_Float16)rb[i].w;
        }
        __syncthreads();

        if (kk + 32 < N_HID) {
#pragma unroll
            for (int i = 0; i < 2; ++i) {
                ap[i] += 32;
                ra[i] = *reinterpret_cast<const v8h*>(ap[i]);
            }
#pragma unroll
            for (int i = 0; i < 4; ++i) {
                bp[i] += (size_t)32 * N_EMBD;
                rb[i] = *reinterpret_cast<const float4*>(bp[i]);
            }
            if (kk + 64 < N_HID)
                __builtin_prefetch(bp[tid & 3] + (size_t)32 * N_EMBD, 0, 1);
        }

        F16Frag af[4], bf[2];
#pragma unroll
        for (int mt = 0; mt < 4; ++mt) {
            const _Float16* p = &sA[(wr + mt * 16 + l15) * LDP + kro];
            af[mt].q[0] = *(const uint4*)p;
            af[mt].q[1] = *(const uint4*)(p + 16);
        }
#pragma unroll
        for (int nt = 0; nt < 2; ++nt) {
            const _Float16* p = &sB[(wc + nt * 16 + l15) * LDP + kro];
            bf[nt].q[0] = *(const uint4*)p;
            bf[nt].q[1] = *(const uint4*)(p + 16);
        }
#pragma unroll
        for (int mt = 0; mt < 4; ++mt)
#pragma unroll
            for (int nt = 0; nt < 2; ++nt)
                acc[mt][nt] = __builtin_amdgcn_wmma_f32_16x16x32_f16(
                    false, af[mt].h, false, bf[nt].h, (short)0, acc[mt][nt], false, false);
    }

#pragma unroll
    for (int mt = 0; mt < 4; ++mt)
#pragma unroll
        for (int nt = 0; nt < 2; ++nt) {
            int col = colbase + wc + nt * 16 + l15;
            float bias = B[col];
#pragma unroll
            for (int i = 0; i < 8; ++i) {
                int row = wr + mt * 16 + mlo + i;
                if (row < rowsv)
                    Yg[(size_t)(slotbase + row) * N_EMBD + col] = acc[mt][nt][i] + bias;
            }
        }
}

// ---------------- combine: out[n] = w0*Y[s0] + w1*Y[s1] (atomic-free) ----------------
__global__ __launch_bounds__(256) void moe_combine(const float* __restrict__ Yg,
                                                   const int* __restrict__ slots,
                                                   const float* __restrict__ ewt,
                                                   float* __restrict__ out)
{
    int idx = blockIdx.x * 256 + threadIdx.x;
    int n = idx >> 10, c = idx & 1023;
    int s0 = slots[2 * n], s1 = slots[2 * n + 1];
    float w0 = ewt[2 * n], w1 = ewt[2 * n + 1];
    out[idx] = w0 * Yg[(size_t)s0 * N_EMBD + c] + w1 * Yg[(size_t)s1 * N_EMBD + c];
}

// ---------------- launch ----------------
extern "C" void kernel_launch(void* const* d_in, const int* in_sizes, int n_in,
                              void* d_out, int out_size, void* d_ws, size_t ws_size,
                              hipStream_t stream)
{
    (void)in_sizes; (void)n_in; (void)out_size; (void)ws_size;
    const float* x  = (const float*)d_in[0];
    const float* gw = (const float*)d_in[1];
    const float* w1 = (const float*)d_in[2];
    const float* b1 = (const float*)d_in[3];
    const float* w2 = (const float*)d_in[4];
    const float* b2 = (const float*)d_in[5];
    float* out = (float*)d_out;

    char* base = (char*)d_ws;
    size_t o = 0;
    auto take = [&](size_t bytes) -> char* {
        char* p = base + o;
        o += (bytes + 255) & ~(size_t)255;
        return p;
    };
    int*      eidx   = (int*)  take((size_t)N_TOKENS * 2 * sizeof(int));
    float*    ewt    = (float*)take((size_t)N_TOKENS * 2 * sizeof(float));
    int*      slots  = (int*)  take((size_t)N_TOKENS * 2 * sizeof(int));
    int*      counts = (int*)  take(N_EXP * sizeof(int));
    int*      offs   = (int*)  take((N_EXP + 1) * sizeof(int));
    int*      cursor = (int*)  take(N_EXP * sizeof(int));
    int*      btok   = (int*)  take((size_t)MAX_SLOTS * sizeof(int));
    _Float16* H      = (_Float16*)take((size_t)MAX_SLOTS * N_HID * sizeof(_Float16));  // 64 MB
    float*    Yg     = (float*)   take((size_t)MAX_SLOTS * N_EMBD * sizeof(float));    // 32 MB

    moe_gate   <<<N_TOKENS / 8, 256, 0, stream>>>(x, gw, eidx, ewt);
    moe_zero   <<<1, 32, 0, stream>>>(counts);
    moe_count  <<<N_TOKENS / 256, 256, 0, stream>>>(eidx, counts);
    moe_scan   <<<1, 32, 0, stream>>>(counts, offs, cursor);
    moe_scatter<<<N_TOKENS / 256, 256, 0, stream>>>(eidx, cursor, btok, slots);

    // Over-launched grids (static shape for graph capture); blocks early-exit
    // past each expert's routed row count read from device memory.
    moe_ffn1<<<dim3(N_HID / 128,  N_TOKENS / 128, N_EXP), 256, 0, stream>>>(
        x, w1, b1, counts, offs, btok, H);
    moe_ffn2<<<dim3(N_EMBD / 128, N_TOKENS / 128, N_EXP), 256, 0, stream>>>(
        H, w2, b2, counts, offs, Yg);
    moe_combine<<<(N_TOKENS * N_EMBD) / 256, 256, 0, stream>>>(Yg, slots, ewt, out);
}
